// BitColumnParallelLinear_18580028523110
// MI455X (gfx1250) — matrix operand order repaired and checked by reading, not compiled, not run on hardware
//
#include <hip/hip_runtime.h>
#include <hip/hip_bf16.h>

// y[b,s,o] = sum_k x[b,s,k] * sign(W[o,k]) + b[o]
// GEMM view: M=8192 (4*2048), K=4096, N=16384; out row-major MxN (== b,s,o).

#define M_DIM 8192
#define K_DIM 4096
#define N_DIM 16384

#define KB 64             // K-depth staged per pipeline stage (2 WMMA k-steps)
#define LDS_STRIDE 72     // KB + 8 pad halves = 144B row stride -> conflict-free b128 reads
#define NSTAGES (K_DIM / KB)  // 64

typedef __attribute__((ext_vector_type(16))) _Float16 v16h;
typedef __attribute__((ext_vector_type(8)))  _Float16 v8h;
typedef __attribute__((ext_vector_type(4)))  _Float16 v4h;
typedef __attribute__((ext_vector_type(8)))  float    v8f;
typedef __attribute__((ext_vector_type(4)))  float    v4f;

// ---- W (f32, N x K row-major) -> sign(W) in f16 -----------------------------
__global__ __launch_bounds__(256) void cvt_sign_w(const float* __restrict__ W,
                                                  _Float16* __restrict__ wh) {
  size_t i = ((size_t)blockIdx.x * 256 + threadIdx.x) * 4;
  v4f w = *(const v4f*)(W + i);
  v4h s;
#pragma unroll
  for (int j = 0; j < 4; ++j) {
    float v = w[j];
    s[j] = (v > 0.0f) ? (_Float16)1.0f
                      : ((v < 0.0f) ? (_Float16)(-1.0f) : (_Float16)0.0f);
  }
  *(v4h*)(wh + i) = s;
}

// ---- x (f32, M x K row-major) -> f16 ---------------------------------------
__global__ __launch_bounds__(256) void cvt_x(const float* __restrict__ X,
                                             _Float16* __restrict__ xh) {
  size_t i = ((size_t)blockIdx.x * 256 + threadIdx.x) * 4;
  v4f v = *(const v4f*)(X + i);
  v4h h;
#pragma unroll
  for (int j = 0; j < 4; ++j) h[j] = (_Float16)v[j];
  *(v4h*)(xh + i) = h;
}

// Async global -> LDS 16-byte copy (per-lane), tracked by ASYNCcnt.
#define ASYNC_B128(ldsOff, gptr)                                         \
  asm volatile("global_load_async_to_lds_b128 %0, %1, off"               \
               :: "v"(ldsOff), "v"((unsigned long long)(size_t)(gptr))   \
               : "memory")

// ---- WMMA GEMM with async-LDS double-buffered pipeline ----------------------
// Block: 128(M) x 128(N), 256 threads = 8 waves in a 4(M) x 2(N) grid.
// Wave: 32(M) x 64(N) = 2 x 4 tiles of 16x16; stage = K-depth 64 = two k-steps
// of v_wmma_f32_16x16x32_f16 (16 WMMAs between barrier pairs).
//
// Stage s: A tile 128x64 f16 + B tile 128x64 f16 staged in LDS (row-major,
// stride 72 halves). Each thread issues 8 global_load_async_to_lds_b128 per
// stage (4 A + 4 B). Steady-state loop is branch-free (final stage peeled):
// issue stage s+1, s_wait_asynccnt 8 retires stage s (in-order), barrier,
// compute, barrier.
__global__ __launch_bounds__(256) void bitlinear_wmma(
    const _Float16* __restrict__ xh,   // M x K f16
    const _Float16* __restrict__ wh,   // N x K f16 (sign values)
    const float*    __restrict__ bias, // N f32
    float*          __restrict__ out)  // M x N f32
{
  __shared__ _Float16 As[2][128 * LDS_STRIDE];  // 2 x 18432 B
  __shared__ _Float16 Bs[2][128 * LDS_STRIDE];  // 2 x 18432 B

  const int tid  = threadIdx.x;
  const int lane = tid & 31;
  const int wave = tid >> 5;
  const int wm   = wave & 3;   // 0..3 along M
  const int wn   = wave >> 2;  // 0..1 along N
  const int lmod = lane & 15;

  const int mBlock = blockIdx.y * 128;
  const int nBlock = blockIdx.x * 128;

  // ---- per-thread async-copy assignment: chunk = 16B (8 halves).
  // Tile = 128 rows x 64 halves = 8 chunks/row -> 1024 chunks -> 4 per thread,
  // at rows (tid>>3) + 32*i, column (tid&7)*8 halves.
  const int crow = tid >> 3;
  const int ccol = (tid & 7) * 8;

  const _Float16* pA[4];
  const _Float16* pB[4];
  unsigned lA[4], lB[4];
#pragma unroll
  for (int i = 0; i < 4; ++i) {
    const int r = crow + 32 * i;
    pA[i] = xh + (size_t)(mBlock + r) * K_DIM + ccol;
    pB[i] = wh + (size_t)(nBlock + r) * K_DIM + ccol;
    lA[i] = (unsigned)(size_t)&As[0][0] + (unsigned)(r * LDS_STRIDE + ccol) * 2;
    lB[i] = (unsigned)(size_t)&Bs[0][0] + (unsigned)(r * LDS_STRIDE + ccol) * 2;
  }
  const unsigned bufStride = 128 * LDS_STRIDE * sizeof(_Float16);  // 18432 B

  const int kOffA = (lane < 16) ? 0 : 8;
  const int kOffB = (lane < 16) ? 0 : 8;

  v8f acc[2][4] = {};

  // Issue one stage's copies into buffer `bo`, then advance the global ptrs.
  auto issueStage = [&](unsigned bo) {
#pragma unroll
    for (int i = 0; i < 4; ++i) {
      __builtin_prefetch(pB[i] + 8 * KB, 0, 0);  // warm L2 on the W stream
      ASYNC_B128(lA[i] + bo, pA[i]);
      ASYNC_B128(lB[i] + bo, pB[i]);
      pA[i] += KB;
      pB[i] += KB;
    }
  };

  // 16 WMMAs on one staged buffer (two k-steps of 32).
  auto computeStage = [&](int buf) {
#pragma unroll
    for (int kk = 0; kk < KB; kk += 32) {
      const _Float16* aBase = &As[buf][(wm * 32 + lmod) * LDS_STRIDE + kk + kOffA];
      v8h a0lo = *(const v8h*)(aBase);
      v8h a0hi = *(const v8h*)(aBase + 16);
      v8h a1lo = *(const v8h*)(aBase + 16 * LDS_STRIDE);
      v8h a1hi = *(const v8h*)(aBase + 16 * LDS_STRIDE + 16);
      v16h a0 = __builtin_shufflevector(a0lo, a0hi, 0, 1, 2, 3, 4, 5, 6, 7,
                                        8, 9, 10, 11, 12, 13, 14, 15);
      v16h a1 = __builtin_shufflevector(a1lo, a1hi, 0, 1, 2, 3, 4, 5, 6, 7,
                                        8, 9, 10, 11, 12, 13, 14, 15);
#pragma unroll
      for (int j = 0; j < 4; ++j) {
        const _Float16* bBase =
            &Bs[buf][(wn * 64 + j * 16 + lmod) * LDS_STRIDE + kk + kOffB];
        v8h blo = *(const v8h*)(bBase);
        v8h bhi = *(const v8h*)(bBase + 16);
        v16h b = __builtin_shufflevector(blo, bhi, 0, 1, 2, 3, 4, 5, 6, 7,
                                         8, 9, 10, 11, 12, 13, 14, 15);
        acc[0][j] = __builtin_amdgcn_wmma_f32_16x16x32_f16(
            false, a0, false, b, (short)0, acc[0][j], false, false);
        acc[1][j] = __builtin_amdgcn_wmma_f32_16x16x32_f16(
            false, a1, false, b, (short)0, acc[1][j], false, false);
      }
    }
  };

  // Prologue: stage 0 -> buffer 0.
  issueStage(0u);

  // Steady state: branch-free body (no in-loop tail special case).
  for (int s = 0; s < NSTAGES - 1; ++s) {
    issueStage((unsigned)((s + 1) & 1) * bufStride);
    asm volatile("s_wait_asynccnt 0x8" ::: "memory");  // stage s retired
    __syncthreads();
    computeStage(s & 1);
    __syncthreads();
  }

  // Peeled final stage.
  asm volatile("s_wait_asynccnt 0x0" ::: "memory");
  __syncthreads();
  computeStage((NSTAGES - 1) & 1);

  // C/D layout: VGPR r -> row mt + r + (lane<16 ? 0 : 8), col nt + lane%16.
  const int rowOff = (lane < 16) ? 0 : 8;
#pragma unroll
  for (int i = 0; i < 2; ++i) {
    const int mt = mBlock + wm * 32 + i * 16;
#pragma unroll
    for (int j = 0; j < 4; ++j) {
      const int col = nBlock + wn * 64 + j * 16 + lmod;
      const float bv = bias[col];
#pragma unroll
      for (int r = 0; r < 8; ++r) {
        out[(size_t)(mt + r + rowOff) * N_DIM + col] = acc[i][j][r] + bv;
      }
    }
  }
}

extern "C" void kernel_launch(void* const* d_in, const int* in_sizes, int n_in,
                              void* d_out, int out_size, void* d_ws, size_t ws_size,
                              hipStream_t stream) {
  const float* x = (const float*)d_in[0];  // 4*2048*4096 f32
  const float* W = (const float*)d_in[1];  // 16384*4096 f32
  const float* b = (const float*)d_in[2];  // 16384 f32
  float* out = (float*)d_out;              // 8192*16384 f32

  _Float16* wh = (_Float16*)d_ws;                                        // 134 MB
  _Float16* xh = (_Float16*)((char*)d_ws +
                             (size_t)N_DIM * K_DIM * sizeof(_Float16));  // +67 MB

  cvt_sign_w<<<(unsigned)((size_t)N_DIM * K_DIM / 4 / 256), 256, 0, stream>>>(W, wh);
  cvt_x<<<(unsigned)((size_t)M_DIM * K_DIM / 4 / 256), 256, 0, stream>>>(x, xh);

  dim3 grid(N_DIM / 128, M_DIM / 128);  // (128, 64)
  bitlinear_wmma<<<grid, 256, 0, stream>>>(xh, wh, b, out);
}
